// my_test_model_9783935500661
// MI455X (gfx1250) — matrix operand order repaired and checked by reading, not compiled, not run on hardware
//
#include <hip/hip_runtime.h>
#include <hip/hip_bf16.h>

#define BATCH 4096

typedef __attribute__((ext_vector_type(16))) _Float16 v16h;
typedef __attribute__((ext_vector_type(8)))  _Float16 v8h;
typedef __attribute__((ext_vector_type(8)))  float    v8f;
typedef __attribute__((ext_vector_type(4)))  int      v4i;

// ---------------------------------------------------------------------------
// WMMA fragment helpers (CDNA5 v_wmma_f32_16x16x32_f16, wave32).
// A (16-bit 16x32, ISA 7.12.2): lane L holds row M = L%16;
//   halves 0-7  = K [kb .. kb+7],  halves 8-15 = K [kb+16 .. kb+23],
//   where kb = (L>=16 ? 8 : 0).  => two aligned 16-byte loads per lane.
// B: lane L holds col N = L%16; halves 0-7 = K [kb .. kb+7], halves 8-15 = K [kb+16 .. kb+23] (as A).
//   Weights are pre-swizzled into this exact order => one 32B load per lane.
// C/D f32: lane L holds col N = L%16; VGPR v holds row M = v + (L>=16?8:0).
// ---------------------------------------------------------------------------

static __device__ __forceinline__ v8f wmma_f32(v16h a, v16h b, v8f c) {
  return __builtin_amdgcn_wmma_f32_16x16x32_f16(false, a, false, b,
                                                (short)0, c, false, false);
}

// Full 16x32 A fragment from row-major f16 [rows][ldk], ldk mult of 16,
// rows 16B aligned, koff mult of 32. Two b128 loads.
static __device__ __forceinline__ v16h load_a_frag32(const _Float16* A,
                                                     int row0, int ldk,
                                                     int koff) {
  int lane = threadIdx.x & 31;
  const _Float16* p =
      A + (size_t)(row0 + (lane & 15)) * ldk + koff + ((lane & 16) ? 8 : 0);
  v8h lo = *(const v8h*)p;
  v8h hi = *(const v8h*)(p + 16);
  return __builtin_shufflevector(lo, hi, 0, 1, 2, 3, 4, 5, 6, 7, 8, 9, 10, 11,
                                 12, 13, 14, 15);
}

// A fragment when K <= 16 (row padded to 16 halves): upper half zero.
static __device__ __forceinline__ v16h load_a_frag16(const _Float16* A,
                                                     int row0, int ldk) {
  int lane = threadIdx.x & 31;
  const _Float16* p =
      A + (size_t)(row0 + (lane & 15)) * ldk + ((lane & 16) ? 8 : 0);
  v8h lo = *(const v8h*)p;
  v8h hi = {};
  return __builtin_shufflevector(lo, hi, 0, 1, 2, 3, 4, 5, 6, 7, 8, 9, 10, 11,
                                 12, 13, 14, 15);
}

// Pre-swizzled B fragment: [nt][ks][lane][16 halves] contiguous.
static __device__ __forceinline__ v16h load_b_swz(const _Float16* Bswz, int nt,
                                                  int ks, int KStot) {
  int lane = threadIdx.x & 31;
  const _Float16* p = Bswz + (((size_t)nt * KStot + ks) * 32 + lane) * 16;
  return *(const v16h*)p;
}

// Store f32 C/D tile to row-major [rows][ldn]. No guards: all destinations
// are tile-padded, so the 8 stores become immediate-offset stores off one
// base address (ldn is a compile-time constant at every call site).
static __device__ __forceinline__ void wmma_store_c(float* D, int ldn, int row0,
                                                    int n0, v8f acc) {
  int lane  = threadIdx.x & 31;
  int n     = n0 + (lane & 15);
  int mbase = (lane & 16) ? 8 : 0;
  float* p = D + (size_t)(row0 + mbase) * ldn + n;
#pragma unroll
  for (int v = 0; v < 8; ++v) p[(size_t)v * ldn] = acc[v];
}

static __device__ __forceinline__ float sigm(float x) {
  return 1.0f / (1.0f + __expf(-x));
}

// ---------------------------------------------------------------------------
// Weight prep: src f32 [N][K] row-major -> f16 fragments in per-lane order.
// dst element idx = ((nt*KS + ks)*32 + lane)*16 + h,
//   n = nt*16 + lane%16,  k = ks*32 + (lane>=16 ? 8 : 0) + (h<8 ? h : h+8).
// ---------------------------------------------------------------------------
__global__ void swizzle_b_f16(const float* __restrict__ W,
                              _Float16* __restrict__ dst, int N, int K, int NT,
                              int KS) {
  int idx = blockIdx.x * blockDim.x + threadIdx.x;
  if (idx >= NT * KS * 512) return;
  int h    = idx & 15;
  int lane = (idx >> 4) & 31;
  int ks   = (idx >> 9) % KS;
  int nt   = (idx >> 9) / KS;
  int n = nt * 16 + (lane & 15);
  int k = ks * 32 + ((lane & 16) ? 8 : 0) + (h < 8 ? h : h + 8);
  float val = (n < N && k < K) ? W[(size_t)n * K + k] : 0.f;
  dst[idx] = (_Float16)val;
}

// ---------------------------------------------------------------------------
// conv1 (1->10, 5x5 VALID on 28x28) + 2x2 maxpool + tanh.
// out: a1h f16 [144 cells][B][16]  (channels 10..15 zero-padded)
// ---------------------------------------------------------------------------
__global__ void conv1_pool_tanh(const float* __restrict__ x,
                                const float* __restrict__ w,
                                const float* __restrict__ bias,
                                _Float16* __restrict__ out) {
  int idx = blockIdx.x * blockDim.x + threadIdx.x;  // b*144 + cell
  if (idx >= BATCH * 144) return;
  int b = idx / 144, cell = idx % 144;
  int ci = cell / 12, cj = cell % 12;
  const float* img = x + (size_t)b * 784;
  _Float16* op = out + (((size_t)cell * BATCH + b) << 4);
  for (int oc = 0; oc < 16; ++oc) {
    _Float16 val = (_Float16)0.f;
    if (oc < 10) {
      float m = -1e30f;
      for (int di = 0; di < 2; ++di)
        for (int dj = 0; dj < 2; ++dj) {
          int oi = 2 * ci + di, oj = 2 * cj + dj;
          float s = bias[oc];
          for (int ki = 0; ki < 5; ++ki)
            for (int kj = 0; kj < 5; ++kj)
              s += img[(oi + ki) * 28 + (oj + kj)] * w[oc * 25 + ki * 5 + kj];
          m = fmaxf(m, s);
        }
      val = (_Float16)tanhf(m);
    }
    op[oc] = val;
  }
}

// ---------------------------------------------------------------------------
// One anti-diagonal of the 2D-LSTM wavefront. One wave per block computes a
// 16(batch) x (5*HID) gate tile with WMMA (A fragments hoisted out of the
// N-tile loop), stages G in LDS, applies gates.
// xin:  f16 [H*W][B][CINP]   Bx/Bl/Bu: pre-swizzled fragments
// cbuf: f32 [H*W][B][HID]    hbuf: f16 [H*W][B][HIDP] (zero-padded)
// ---------------------------------------------------------------------------
template <int HID, int CIN>
__global__ __launch_bounds__(32) void mdlstm_diag(
    const _Float16* __restrict__ xin, const _Float16* __restrict__ Bx,
    const _Float16* __restrict__ Bl, const _Float16* __restrict__ Bu,
    const float* __restrict__ bias, float* __restrict__ cbuf,
    _Float16* __restrict__ hbuf, int H, int W, int d) {
  constexpr int GD   = 5 * HID;
  constexpr int NT   = (GD + 15) / 16;
  constexpr int GDP  = NT * 16;
  constexpr int HIDP = ((HID + 31) / 32) * 32;   // 32 or 64
  constexpr int CINP = ((CIN + 15) / 16) * 16;   // 16 or 32
  constexpr int KS   = HIDP / 32;                // 1 or 2
  __shared__ float Gs[16 * GDP];

  int lane = threadIdx.x & 31;
  int b0   = blockIdx.x * 16;
  int i_lo = d - (W - 1); if (i_lo < 0) i_lo = 0;
  int i    = i_lo + blockIdx.y;
  int j    = d - i;
  int cell = i * W + j;

  const _Float16* Ax = xin + (size_t)cell * BATCH * CINP;
  const _Float16* Al = (j > 0) ? hbuf + (size_t)(cell - 1) * BATCH * HIDP : nullptr;
  const _Float16* Au = (i > 0) ? hbuf + (size_t)(cell - W) * BATCH * HIDP : nullptr;

  // A fragments are N-invariant: load once.
  v16h ax;
  if constexpr (CINP == 16) ax = load_a_frag16(Ax, b0, CINP);
  else                      ax = load_a_frag32(Ax, b0, CINP, 0);
  v16h al[KS], au[KS];
#pragma unroll
  for (int ks = 0; ks < KS; ++ks) {
    al[ks] = (Al != nullptr) ? load_a_frag32(Al, b0, HIDP, ks * 32) : (v16h){};
    au[ks] = (Au != nullptr) ? load_a_frag32(Au, b0, HIDP, ks * 32) : (v16h){};
  }
  bool hasL = (Al != nullptr), hasU = (Au != nullptr);

  for (int nt = 0; nt < NT; ++nt) {
    int   n0 = nt * 16;
    int   n  = n0 + (lane & 15);
    float bv = (n < GD) ? bias[n] : 0.f;
    v8f acc = {bv, bv, bv, bv, bv, bv, bv, bv};
    acc = wmma_f32(ax, load_b_swz(Bx, nt, 0, 1), acc);
    if (hasL) {
#pragma unroll
      for (int ks = 0; ks < KS; ++ks)
        acc = wmma_f32(al[ks], load_b_swz(Bl, nt, ks, KS), acc);
    }
    if (hasU) {
#pragma unroll
      for (int ks = 0; ks < KS; ++ks)
        acc = wmma_f32(au[ks], load_b_swz(Bu, nt, ks, KS), acc);
    }
    wmma_store_c(Gs, GDP, 0, n0, acc);   // full padded width, no guards
  }
  __syncthreads();

  float*       cc = cbuf + (size_t)cell * BATCH * HID;
  _Float16*    hc = hbuf + (size_t)cell * BATCH * HIDP;
  const float* cl = (j > 0) ? cbuf + (size_t)(cell - 1) * BATCH * HID : nullptr;
  const float* cu = (i > 0) ? cbuf + (size_t)(cell - W) * BATCH * HID : nullptr;

  for (int t = lane; t < 16 * HIDP; t += 32) {
    int row = t / HIDP;          // HIDP power of two -> shift
    int hh  = t & (HIDP - 1);
    int b   = b0 + row;
    if (hh < HID) {
      float gi  = Gs[row * GDP + hh];
      float gf1 = Gs[row * GDP + HID + hh];
      float gf2 = Gs[row * GDP + 2 * HID + hh];
      float go  = Gs[row * GDP + 3 * HID + hh];
      float gg  = Gs[row * GDP + 4 * HID + hh];
      float clv = cl ? cl[(size_t)b * HID + hh] : 0.f;
      float cuv = cu ? cu[(size_t)b * HID + hh] : 0.f;
      float c   = sigm(gf1) * clv + sigm(gf2) * cuv + sigm(gi) * tanhf(gg);
      float hv  = sigm(go) * tanhf(c);
      cc[(size_t)b * HID + hh]  = c;
      hc[(size_t)b * HIDP + hh] = (_Float16)hv;
    } else {
      hc[(size_t)b * HIDP + hh] = (_Float16)0.f;  // keep padding zero
    }
  }
}

// ---------------------------------------------------------------------------
// conv2 as im2col GEMM: M = B*64 (pre-pool 8x8), N = 30 (2 tiles, both per
// block), K = 500 pad 512. Gather offsets come from an LDS table:
//   addr = basecell*131072 + b*32 + (dcell*131072 + ic)   [all shifts, B=4096]
// k >= 500 points at a padded zero slot (ic in 20..31, dcell 0) so every
// gather is an unconditional u16 load -- no sentinel branches, full MLP.
// h1h f16 [144][B][32], W2T pre-swizzled [2][16][512], y f32 [B*64][32]
// ---------------------------------------------------------------------------
__global__ __launch_bounds__(32) void conv2_gemm(const _Float16* __restrict__ h1h,
                                                 const _Float16* __restrict__ W2T,
                                                 const float* __restrict__ bias,
                                                 float* __restrict__ y) {
  __shared__ int tbl[512];
  int lane = threadIdx.x & 31;
  for (int k = lane; k < 512; k += 32) {
    int v;
    if (k < 500) {
      int ic = k / 25, rem = k % 25, ki = rem / 5, kj = rem % 5;
      v = (ki * 12 + kj) * 131072 + ic;
    } else {
      v = 20 + (k - 500);  // padded zero half within row 0 offset
    }
    tbl[k] = v;
  }
  __syncthreads();

  int row0 = blockIdx.x * 16;
  int r    = row0 + (lane & 15);
  int b    = r >> 6, pos = r & 63, oi = pos >> 3, oj = pos & 7;
  const _Float16* hrow =
      h1h + (size_t)(oi * 12 + oj) * 131072 + ((size_t)b << 5);

  int n = lane & 15;
  float bv0 = bias[n];                       // n-tile 0: n < 16 <= 30
  float bv1 = (16 + n < 30) ? bias[16 + n] : 0.f;
  v8f acc0 = {bv0, bv0, bv0, bv0, bv0, bv0, bv0, bv0};
  v8f acc1 = {bv1, bv1, bv1, bv1, bv1, bv1, bv1, bv1};

  int khalf = (lane & 16) ? 8 : 0;
  for (int ks = 0; ks < 16; ++ks) {
    int base = ks * 32 + khalf;              // 32B-aligned LDS offset
    v4i t0 = *(const v4i*)&tbl[base];
    v4i t1 = *(const v4i*)&tbl[base + 4];
    v4i t2 = *(const v4i*)&tbl[base + 16];
    v4i t3 = *(const v4i*)&tbl[base + 20];
    v16h a;
#pragma unroll
    for (int e = 0; e < 4; ++e) {
      a[e]      = hrow[t0[e]];
      a[4 + e]  = hrow[t1[e]];
      a[8 + e]  = hrow[t2[e]];
      a[12 + e] = hrow[t3[e]];
    }
    acc0 = wmma_f32(a, load_b_swz(W2T, 0, ks, 16), acc0);
    acc1 = wmma_f32(a, load_b_swz(W2T, 1, ks, 16), acc1);
  }
  wmma_store_c(y, 32, row0, 0,  acc0);
  wmma_store_c(y, 32, row0, 16, acc1);
}

// 2x2 maxpool + tanh. y f32 [B*64][32] -> a2h f16 [16][B][32] (zero-padded)
__global__ void pool2_tanh(const float* __restrict__ y,
                           _Float16* __restrict__ a2h) {
  int idx = blockIdx.x * blockDim.x + threadIdx.x;  // b*512 + cell*32 + oc
  if (idx >= BATCH * 16 * 32) return;
  int oc = idx & 31, cell = (idx >> 5) & 15, b = idx >> 9;
  _Float16 val = (_Float16)0.f;
  if (oc < 30) {
    int ci = cell >> 2, cj = cell & 3;
    float m = -1e30f;
    for (int di = 0; di < 2; ++di)
      for (int dj = 0; dj < 2; ++dj) {
        int row = b * 64 + (2 * ci + di) * 8 + (2 * cj + dj);
        m = fmaxf(m, y[((size_t)row << 5) + oc]);
      }
    val = (_Float16)tanhf(m);
  }
  a2h[(((size_t)cell * BATCH + b) << 5) + oc] = val;
}

// h2h f16 [16][B][64] -> h2f f16 [B][640], k = c*16 + cell (reference order)
__global__ void repack_h2(const _Float16* __restrict__ h2h,
                          _Float16* __restrict__ h2f) {
  int idx = blockIdx.x * blockDim.x + threadIdx.x;
  if (idx >= BATCH * 640) return;
  int b = idx / 640, k = idx % 640;
  int c = k >> 4, cell = k & 15;
  h2f[idx] = h2h[(((size_t)cell * BATCH + b) << 6) + c];
}

// fc1: z1[B][112 padded] = h2f[B][640] @ fc1_w^T + b
__global__ __launch_bounds__(32) void fc1_gemm(const _Float16* __restrict__ h2f,
                                               const _Float16* __restrict__ F1T,
                                               const float* __restrict__ bias,
                                               float* __restrict__ z1) {
  int lane = threadIdx.x & 31;
  int b0 = blockIdx.x * 16;
  int n0 = blockIdx.y * 16;
  int n  = n0 + (lane & 15);
  float bv = (n < 100) ? bias[n] : 0.f;
  v8f acc = {bv, bv, bv, bv, bv, bv, bv, bv};
  for (int ks = 0; ks < 20; ++ks) {
    v16h a  = load_a_frag32(h2f, b0, 640, ks * 32);
    v16h bm = load_b_swz(F1T, blockIdx.y, ks, 20);
    acc = wmma_f32(a, bm, acc);
  }
  wmma_store_c(z1, 112, b0, n0, acc);   // padded ld, no guards
}

// tanh(fc1) -> fc2 -> softmax, one thread per batch row. z1 ld = 112.
__global__ void fc2_softmax(const float* __restrict__ z1,
                            const float* __restrict__ w2,
                            const float* __restrict__ b2,
                            float* __restrict__ out) {
  int b = blockIdx.x * blockDim.x + threadIdx.x;
  if (b >= BATCH) return;
  float hbuf[100];
  for (int k = 0; k < 100; ++k) hbuf[k] = tanhf(z1[(size_t)b * 112 + k]);
  float o[10];
  float mx = -1e30f;
  for (int g = 0; g < 10; ++g) {
    float s = b2[g];
    for (int k = 0; k < 100; ++k) s += hbuf[k] * w2[g * 100 + k];
    o[g] = s;
    mx = fmaxf(mx, s);
  }
  float den = 0.f;
  for (int g = 0; g < 10; ++g) { o[g] = __expf(o[g] - mx); den += o[g]; }
  for (int g = 0; g < 10; ++g) out[(size_t)b * 10 + g] = o[g] / den;
}

// ---------------------------------------------------------------------------
// Host-side orchestration
// ---------------------------------------------------------------------------
static inline int imax_(int a, int b) { return a > b ? a : b; }
static inline int imin_(int a, int b) { return a < b ? a : b; }

extern "C" void kernel_launch(void* const* d_in, const int* in_sizes, int n_in,
                              void* d_out, int out_size, void* d_ws,
                              size_t ws_size, hipStream_t stream) {
  const float* x       = (const float*)d_in[0];
  const float* conv1_w = (const float*)d_in[1];
  const float* conv1_b = (const float*)d_in[2];
  const float* m1_Wx   = (const float*)d_in[3];
  const float* m1_Whl  = (const float*)d_in[4];
  const float* m1_Whu  = (const float*)d_in[5];
  const float* m1_b    = (const float*)d_in[6];
  const float* conv2_w = (const float*)d_in[7];
  const float* conv2_b = (const float*)d_in[8];
  const float* m2_Wx   = (const float*)d_in[9];
  const float* m2_Whl  = (const float*)d_in[10];
  const float* m2_Whu  = (const float*)d_in[11];
  const float* m2_b    = (const float*)d_in[12];
  const float* fc1_w   = (const float*)d_in[13];
  const float* fc1_b   = (const float*)d_in[14];
  const float* fc2_w   = (const float*)d_in[15];
  const float* fc2_b   = (const float*)d_in[16];
  float* out = (float*)d_out;
  (void)in_sizes; (void)n_in; (void)out_size;

  char* ws = (char*)d_ws;
  size_t off = 0;
  auto alloc = [&](size_t bytes) -> void* {
    void* p = ws + off;
    off = (off + bytes + 255) & ~(size_t)255;
    return p;
  };
  _Float16* a1h = (_Float16*)alloc((size_t)144 * BATCH * 16 * 2);
  _Float16* h1h = (_Float16*)alloc((size_t)144 * BATCH * 32 * 2);
  float*    c1  = (float*)   alloc((size_t)144 * BATCH * 20 * 4);
  float*    y2  = (float*)   alloc((size_t)BATCH * 64 * 32 * 4);
  _Float16* a2h = (_Float16*)alloc((size_t)16 * BATCH * 32 * 2);
  _Float16* h2h = (_Float16*)alloc((size_t)16 * BATCH * 64 * 2);
  float*    c2  = (float*)   alloc((size_t)16 * BATCH * 40 * 4);
  _Float16* h2f = (_Float16*)alloc((size_t)BATCH * 640 * 2);
  float*    z1  = (float*)   alloc((size_t)BATCH * 112 * 4);
  _Float16* Bx1 = (_Float16*)alloc((size_t)7 * 1 * 512 * 2);
  _Float16* Bl1 = (_Float16*)alloc((size_t)7 * 1 * 512 * 2);
  _Float16* Bu1 = (_Float16*)alloc((size_t)7 * 1 * 512 * 2);
  _Float16* Bx2 = (_Float16*)alloc((size_t)13 * 1 * 512 * 2);
  _Float16* Bl2 = (_Float16*)alloc((size_t)13 * 2 * 512 * 2);
  _Float16* Bu2 = (_Float16*)alloc((size_t)13 * 2 * 512 * 2);
  _Float16* W2T = (_Float16*)alloc((size_t)2 * 16 * 512 * 2);
  _Float16* F1T = (_Float16*)alloc((size_t)7 * 20 * 512 * 2);
  if (off > ws_size) return;  // workspace too small; bail deterministically

  auto SW = [&](const float* src, _Float16* dst, int N, int K, int NT, int KS) {
    int nel = NT * KS * 512;
    swizzle_b_f16<<<(nel + 255) / 256, 256, 0, stream>>>(src, dst, N, K, NT, KS);
  };
  SW(m1_Wx,   Bx1, 100, 10,  7, 1);
  SW(m1_Whl,  Bl1, 100, 20,  7, 1);
  SW(m1_Whu,  Bu1, 100, 20,  7, 1);
  SW(m2_Wx,   Bx2, 200, 30, 13, 1);
  SW(m2_Whl,  Bl2, 200, 40, 13, 2);
  SW(m2_Whu,  Bu2, 200, 40, 13, 2);
  SW(conv2_w, W2T,  30, 500, 2, 16);
  SW(fc1_w,   F1T, 100, 640, 7, 20);

  conv1_pool_tanh<<<(BATCH * 144 + 255) / 256, 256, 0, stream>>>(x, conv1_w,
                                                                 conv1_b, a1h);

  for (int d = 0; d <= 22; ++d) {  // 12x12 wavefront
    int i_lo = imax_(0, d - 11), i_hi = imin_(11, d);
    dim3 grid(BATCH / 16, i_hi - i_lo + 1);
    mdlstm_diag<20, 10><<<grid, 32, 0, stream>>>(a1h, Bx1, Bl1, Bu1, m1_b, c1,
                                                 h1h, 12, 12, d);
  }

  conv2_gemm<<<BATCH * 64 / 16, 32, 0, stream>>>(h1h, W2T, conv2_b, y2);
  pool2_tanh<<<(BATCH * 16 * 32 + 255) / 256, 256, 0, stream>>>(y2, a2h);

  for (int d = 0; d <= 6; ++d) {  // 4x4 wavefront
    int i_lo = imax_(0, d - 3), i_hi = imin_(3, d);
    dim3 grid(BATCH / 16, i_hi - i_lo + 1);
    mdlstm_diag<40, 30><<<grid, 32, 0, stream>>>(a2h, Bx2, Bl2, Bu2, m2_b, c2,
                                                 h2h, 4, 4, d);
  }

  repack_h2<<<(BATCH * 640 + 255) / 256, 256, 0, stream>>>(h2h, h2f);
  fc1_gemm<<<dim3(BATCH / 16, 7), 32, 0, stream>>>(h2f, F1T, fc1_b, z1);
  fc2_softmax<<<(BATCH + 127) / 128, 128, 0, stream>>>(z1, fc2_w, fc2_b, out);
}
